// ContrastiveLoss_29016799051778
// MI455X (gfx1250) — matrix operand order repaired and checked by reading, not compiled, not run on hardware
//
#include <hip/hip_runtime.h>
#include <hip/hip_bf16.h>

typedef __attribute__((ext_vector_type(16))) _Float16 v16h;
typedef __attribute__((ext_vector_type(4)))  _Float16 v4h;
typedef __attribute__((ext_vector_type(8)))  float    v8f;
typedef __attribute__((ext_vector_type(4)))  unsigned int v4u;

#define B_  4
#define N_  4096
#define D_  128
#define R_  (B_ * N_)          /* 16384 rows total */
#define TEMP_INV 10.0f         /* 1 / 0.1 */

// ---------------------------------------------------------------------------
// Kernel 1: L2-normalize each 128-elem row, split into f16 hi/lo planes.
// One wave32 per row: 4 floats per lane, shuffle reduction for the norm.
// ---------------------------------------------------------------------------
__global__ __launch_bounds__(256) void nrm_split_kernel(
    const float* __restrict__ emb,
    _Float16* __restrict__ hi,
    _Float16* __restrict__ lo) {
  const int gid  = blockIdx.x * blockDim.x + threadIdx.x;
  const int row  = gid >> 5;
  const int lane = threadIdx.x & 31;
  if (row >= R_) return;

  const float4 x = ((const float4* __restrict__)(emb + (size_t)row * D_))[lane];
  float ss = x.x * x.x + x.y * x.y + x.z * x.z + x.w * x.w;
#pragma unroll
  for (int m = 16; m >= 1; m >>= 1) ss += __shfl_xor(ss, m, 32);
  const float inv = 1.0f / fmaxf(sqrtf(ss), 1e-12f);

  const float vals[4] = { x.x * inv, x.y * inv, x.z * inv, x.w * inv };
  v4h h, l;
#pragma unroll
  for (int i = 0; i < 4; ++i) {
    const _Float16 hv = (_Float16)vals[i];
    h[i] = hv;
    l[i] = (_Float16)(vals[i] - (float)hv);   // residual for hi/lo split
  }
  const size_t off = (size_t)row * D_ + (size_t)lane * 4;
  *(v4h*)(hi + off) = h;
  *(v4h*)(lo + off) = l;
}

// ---------------------------------------------------------------------------
// Operand loaders matching the CDNA5 16-bit WMMA VGPR layouts (ISA 7.12.2).
// A (16x32, lane m = L%16, sel = L/16): halves 0..7 = K[sel*8 .. sel*8+7],
//                                       halves 8..15 = K[16+sel*8 ..].
// B (32x16, lane n = L%16, sel = L/16): halves 0..7 = K[sel*8 .. sel*8+7],
//                                       halves 8..15 = K[16+sel*8 ..].
// Rows are 256B-strided and 16B-aligned -> global_load_b128 pairs.
// ---------------------------------------------------------------------------
__device__ inline v16h loadA16(const _Float16* __restrict__ row, int kbase, int sel) {
  union { v16h h; v4u u[2]; } r;
  r.u[0] = *(const v4u*)(row + kbase + sel * 8);
  r.u[1] = *(const v4u*)(row + kbase + 16 + sel * 8);
  return r.h;
}
__device__ inline v16h loadB16(const _Float16* __restrict__ row, int kbase, int sel) {
  union { v16h h; v4u u[2]; } r;
  const _Float16* p = row + kbase + sel * 8;
  r.u[0] = *(const v4u*)(p);
  r.u[1] = *(const v4u*)(p + 16);
  return r.h;
}

#define WMMA_F16(a, b, c) \
  __builtin_amdgcn_wmma_f32_16x16x32_f16(false, (a), false, (b), (short)0, (c), false, false)

// ---------------------------------------------------------------------------
// Kernel 2: one wave32 per 16-row tile. A = 16 normalized rows, B = 16
// gathered partner rows as columns. 24 v_wmma_f32_16x16x32_f16 (hi/lo split,
// pos + neg), diagonal extraction, stable softplus, wave reduce, atomic add.
// Grid exactly covers all tiles so EXEC is all-1s at every WMMA.
// ---------------------------------------------------------------------------
__global__ __launch_bounds__(256) void wmma_loss_kernel(
    const _Float16* __restrict__ hi,
    const _Float16* __restrict__ lo,
    const int* __restrict__ posIdx,
    const int* __restrict__ negIdx,
    float* __restrict__ out) {
  const int lane = threadIdx.x & 31;
  const int tile = blockIdx.x * (blockDim.x >> 5) + (threadIdx.x >> 5); // 0..1023
  const int r0   = tile * 16;
  const int b    = r0 / N_;
  const int idx  = lane & 15;   // row (A) / column (B) handled by this lane
  const int sel  = lane >> 4;   // K-chunk selector within the operand layout

  const int jp = posIdx[r0 + idx];
  const int jn = negIdx[r0 + idx];

  const _Float16* aHiRow = hi + (size_t)(r0 + idx) * D_;
  const _Float16* aLoRow = lo + (size_t)(r0 + idx) * D_;
  const size_t bpOff = (size_t)(b * N_ + jp) * D_;
  const size_t bnOff = (size_t)(b * N_ + jn) * D_;

  v8f accP = {};
  v8f accN = {};
#pragma unroll
  for (int kc = 0; kc < 4; ++kc) {
    const int kbase = kc * 32;
    const v16h aH = loadA16(aHiRow, kbase, sel);
    const v16h aL = loadA16(aLoRow, kbase, sel);
    const v16h pH = loadB16(hi + bpOff, kbase, sel);
    const v16h pL = loadB16(lo + bpOff, kbase, sel);
    const v16h nH = loadB16(hi + bnOff, kbase, sel);
    const v16h nL = loadB16(lo + bnOff, kbase, sel);
    // 3-term f16 split: hi*hi + hi*lo + lo*hi  (f32 accumulate)
    accP = WMMA_F16(aH, pH, accP);
    accP = WMMA_F16(aH, pL, accP);
    accP = WMMA_F16(aL, pH, accP);
    accN = WMMA_F16(aH, nH, accN);
    accN = WMMA_F16(aH, nL, accN);
    accN = WMMA_F16(aL, nH, accN);
  }

  // Diagonal of the 16x16 C tile (ISA C-layout): rows 0..7 live in lanes 0..7
  // at VGPR v = lane; rows 8..15 live in lanes 24..31 at VGPR v = lane-24.
  const bool active = (lane < 8) || (lane >= 24);
  const int  v = (lane < 8) ? lane : (lane - 24);
  float dP = accP[0], dN = accN[0];
#pragma unroll
  for (int i = 1; i < 8; ++i) {
    dP = (v == i) ? accP[i] : dP;
    dN = (v == i) ? accN[i] : dN;
  }

  float contrib = 0.0f;
  if (active) {
    const float xP = dP * TEMP_INV;
    const float xN = dN * TEMP_INV;
    // -log_sigmoid(x)  = softplus(-x) = max(-x,0) + log1p(exp(-|x|))
    // -log_sigmoid(-x) = softplus( x)
    const float sp1 = fmaxf(-xP, 0.0f) + log1pf(expf(-fabsf(xP)));
    const float sp2 = fmaxf( xN, 0.0f) + log1pf(expf(-fabsf(xN)));
    contrib = sp1 + sp2;
  }
#pragma unroll
  for (int m = 16; m >= 1; m >>= 1) contrib += __shfl_xor(contrib, m, 32);
  if (lane == 0) atomicAdd(out, contrib * (1.0f / (float)R_));
}

// ---------------------------------------------------------------------------
extern "C" void kernel_launch(void* const* d_in, const int* in_sizes, int n_in,
                              void* d_out, int out_size, void* d_ws, size_t ws_size,
                              hipStream_t stream) {
  const float* emb = (const float*)d_in[0];
  const int*   pos = (const int*)d_in[1];   // harness passes integer inputs as int32
  const int*   neg = (const int*)d_in[2];
  float*       out = (float*)d_out;

  _Float16* hi = (_Float16*)d_ws;                                   // 4 MB
  _Float16* lo = (_Float16*)((char*)d_ws + (size_t)R_ * D_ * 2);    // 4 MB

  (void)in_sizes; (void)n_in; (void)out_size; (void)ws_size;

  hipMemsetAsync(d_out, 0, sizeof(float), stream);
  nrm_split_kernel<<<R_ / 8, 256, 0, stream>>>(emb, hi, lo);
  wmma_loss_kernel<<<(R_ / 16) / 8, 256, 0, stream>>>(hi, lo, pos, neg, out);
}
